// MultiHead_19396072309193
// MI455X (gfx1250) — matrix operand order repaired and checked by reading, not compiled, not run on hardware
//
#include <hip/hip_runtime.h>
#include <hip/hip_bf16.h>

typedef __attribute__((ext_vector_type(16))) __bf16 v16bf;
typedef __attribute__((ext_vector_type(8)))  float  v8f;

constexpr int B = 2, S = 2048, D = 1024, H = 16, DK = 64;
constexpr int M = B * S;              // 4096 rows for all big GEMMs
constexpr float SCALE = 0.125f;       // 1/sqrt(DK)

__device__ __forceinline__ unsigned short f2bf(float f) {
  unsigned u = __float_as_uint(f);
  u += 0x7FFFu + ((u >> 16) & 1u);    // round-to-nearest-even
  return (unsigned short)(u >> 16);
}
__device__ __forceinline__ float bf2f(unsigned short h) {
  return __uint_as_float(((unsigned)h) << 16);
}

union ABReg { unsigned u[8]; v16bf v; };

__device__ __forceinline__ v8f wmma_bf16(const ABReg& a, const ABReg& b, v8f c) {
  return __builtin_amdgcn_wmma_f32_16x16x32_bf16(false, a.v, false, b.v,
                                                 (short)0, c, false, false);
}

// ---------------------------------------------------------------- convert
__global__ void convert_f32_bf16(const float* __restrict__ in,
                                 unsigned short* __restrict__ out, int n) {
  for (int i = blockIdx.x * blockDim.x + threadIdx.x; i < n; i += gridDim.x * blockDim.x)
    out[i] = f2bf(in[i]);
}

// ---------------------------------------------------------------- GEMM
// C[m,n] = sum_k A[m,k] * W[n,k], M=4096, N=K=1024, bf16 in / f32 acc.
// Block (8 waves) computes 128x128. 32-deep k-slabs move through an explicit
// LDS double buffer: loads for slab i+1 issue at the top of iteration i,
// compute reads buf[cur], arrived data is stored into buf[cur^1], and a
// single barrier per iteration separates producers from next consumers —
// no value is live across any barrier.
// mode 0: bf16 out [B,H,S,DK]; mode 1: bf16 out [B,H,DK,S]; mode 2: f32 [M,D]
__global__ void __launch_bounds__(256, 2)
gemm_bf16_wmma(const unsigned short* __restrict__ A,
               const unsigned short* __restrict__ W,
               void* __restrict__ out, int mode) {
  __shared__ __align__(16) unsigned short As[2][128 * 32];  // 2 x 8 KB
  __shared__ __align__(16) unsigned short Ws[2][128 * 32];  // 2 x 8 KB

  const int tid    = threadIdx.x;
  const int lane   = tid & 31;
  const int w      = tid >> 5;
  const int lane16 = lane & 15;
  const int half   = lane >> 4;
  const int m0 = (blockIdx.x >> 3) * 128;   // 32 m-blocks
  const int n0 = (blockIdx.x & 7) * 128;    // 8 n-blocks
  const int mw = (w >> 1) * 32;             // wave subtile: 32 rows
  const int nw = (w & 1) * 64;              //               64 cols

  const int srow  = tid >> 1;               // staging: 2 threads per row
  const int skoff = (tid & 1) * 16;         // 16 elems = 32B each

  v8f acc[2][4] = {};

  const unsigned short* Ag = A + (size_t)(m0 + srow) * D + skoff;
  const unsigned short* Wg = W + (size_t)(n0 + srow) * D + skoff;

  // slab 0 -> buffer 0
#pragma unroll
  for (int i = 0; i < 2; ++i) {
    ((uint4*)&As[0][srow * 32 + skoff])[i] = ((const uint4*)Ag)[i];
    ((uint4*)&Ws[0][srow * 32 + skoff])[i] = ((const uint4*)Wg)[i];
  }
  __syncthreads();

  for (int k0 = 0; k0 < D; k0 += 32) {
    const int  cur  = (k0 >> 5) & 1;
    const bool more = (k0 + 32 < D);

    uint4 ar[2], wr[2];
    if (more) {
#pragma unroll
      for (int i = 0; i < 2; ++i) {
        ar[i] = ((const uint4*)(Ag + k0 + 32))[i];
        wr[i] = ((const uint4*)(Wg + k0 + 32))[i];
      }
    }

    ABReg a[2];
#pragma unroll
    for (int mi = 0; mi < 2; ++mi) {
      int arow = mw + mi * 16 + lane16;
      *(uint4*)&a[mi].u[0] = *(const uint4*)&As[cur][arow * 32 + half * 8];
      *(uint4*)&a[mi].u[4] = *(const uint4*)&As[cur][arow * 32 + half * 8 + 16];
    }
#pragma unroll
    for (int t = 0; t < 4; ++t) {           // B fragment loaded just-in-time
      ABReg b;
      int wrow = nw + t * 16 + lane16;
      *(uint4*)&b.u[0] = *(const uint4*)&Ws[cur][wrow * 32 + half * 8];
      *(uint4*)&b.u[4] = *(const uint4*)&Ws[cur][wrow * 32 + half * 8 + 16];
      acc[0][t] = wmma_bf16(a[0], b, acc[0][t]);
      acc[1][t] = wmma_bf16(a[1], b, acc[1][t]);
    }

    if (more) {
#pragma unroll
      for (int i = 0; i < 2; ++i) {
        ((uint4*)&As[cur ^ 1][srow * 32 + skoff])[i] = ar[i];
        ((uint4*)&Ws[cur ^ 1][srow * 32 + skoff])[i] = wr[i];
      }
    }
    __syncthreads();
  }

#pragma unroll
  for (int mi = 0; mi < 2; ++mi)
#pragma unroll
    for (int t = 0; t < 4; ++t)
#pragma unroll
      for (int r = 0; r < 8; ++r) {
        int m = m0 + mw + mi * 16 + r + 8 * half;
        int n = n0 + nw + t * 16 + lane16;
        float v = acc[mi][t][r];
        if (mode == 2) {
          ((float*)out)[(size_t)m * D + n] = v;
        } else {
          int bb = m >> 11, s = m & (S - 1);
          int hh = n >> 6,  dk = n & 63;
          size_t idx = (mode == 0)
            ? ((size_t)((bb * H + hh) * S + s) * DK + dk)
            : ((size_t)((bb * H + hh) * DK + dk) * S + s);
          ((unsigned short*)out)[idx] = f2bf(v);
        }
      }
}

// ---------------------------------------------------------------- RoPE
__global__ void rope_kernel(unsigned short* __restrict__ q,
                            unsigned short* __restrict__ k,
                            const int* __restrict__ pos) {
  int i = blockIdx.x * blockDim.x + threadIdx.x;     // over B*H*S*(DK/2)
  if (i >= B * H * S * (DK / 2)) return;
  int p2 = i & 31;
  int s  = (i >> 5) & (S - 1);
  int h  = (i >> 16) & (H - 1);
  int b  = i >> 20;
  float position = (float)pos[b * S + s];
  float inv = __expf(-(float)p2 * (9.210340371976184f / 32.0f)); // theta^(-2i/64)
  float sn, c;
  __sincosf(position * inv, &sn, &c);
  size_t base = ((size_t)((b * H + h) * S + s)) * DK + 2 * p2;
  float qe = bf2f(q[base]), qo = bf2f(q[base + 1]);
  q[base]     = f2bf(qe * c - qo * sn);
  q[base + 1] = f2bf(qe * sn + qo * c);
  float ke = bf2f(k[base]), ko = bf2f(k[base + 1]);
  k[base]     = f2bf(ke * c - ko * sn);
  k[base + 1] = f2bf(ke * sn + ko * c);
}

// ---------------------------------------------------------------- attention
// Block = 8 waves = 128 consecutive query rows of one (b,h). Q tile (16 KB)
// staged once; K/V move through an explicit LDS double buffer (one barrier
// per 32-key step, no values live across barriers). Fragments are loaded
// from LDS just-in-time so the persistent set (O 32 + m/l 16) fits ~128 VGPRs.
__global__ void __launch_bounds__(256, 2)
attn_kernel(const unsigned short* __restrict__ Q,
            const unsigned short* __restrict__ K,
            const unsigned short* __restrict__ Vt,
            unsigned short* __restrict__ O) {
  __shared__ __align__(16) unsigned short Qs[128 * 64];     // [qrow][dk] 16 KB
  __shared__ __align__(16) unsigned short Ks[2][32 * 64];   // [key][dk] 2x4 KB
  __shared__ __align__(16) unsigned short Vs[2][64 * 32];   // [dk][key] 2x4 KB
  __shared__ __align__(16) unsigned short pbuf[8][16 * 32]; // per-wave P 8 KB

  const int tid    = threadIdx.x;
  const int lane   = tid & 31;
  const int w      = tid >> 5;
  const int lane16 = lane & 15;
  const int half   = lane >> 4;
  const int qb = blockIdx.x & 15;            // 16 q-blocks of 128 rows
  const int h  = (blockIdx.x >> 4) & (H - 1);
  const int b  = blockIdx.x >> 8;
  const int q0 = qb * 128 + w * 16;          // this wave's 16 query rows

  const unsigned short* Qb = Q  + (size_t)((b * H + h) * S) * DK;
  const unsigned short* Kb = K  + (size_t)((b * H + h) * S) * DK;
  const unsigned short* Vb = Vt + (size_t)((b * H + h) * DK) * S;

  // staging assignments
  const int krow = tid >> 3, koff = (tid & 7) * 8;   // K tile 32x64
  const int vrow = tid >> 2, voff = (tid & 3) * 8;   // V tile 64x32

  // stage the block's Q tile once + slab 0 of K/V into buffer 0
  {
    const int qsrow = tid >> 1;              // 0..127
    const int qsoff = (tid & 1) * 32;        // 32 elems = 64B
    const unsigned short* Qg = Qb + (size_t)(qb * 128 + qsrow) * DK + qsoff;
    ((uint4*)&Qs[qsrow * 64 + qsoff])[0] = ((const uint4*)Qg)[0];
    ((uint4*)&Qs[qsrow * 64 + qsoff])[1] = ((const uint4*)Qg)[1];
    *(uint4*)&Ks[0][krow * 64 + koff] = *(const uint4*)(Kb + (size_t)krow * DK + koff);
    *(uint4*)&Vs[0][vrow * 32 + voff] = *(const uint4*)(Vb + (size_t)vrow * S + voff);
  }
  __syncthreads();

  float mrow[8], lrow[8];
  v8f o[4] = {};
#pragma unroll
  for (int r = 0; r < 8; ++r) { mrow[r] = -1e30f; lrow[r] = 0.f; }

  const int kend   = qb * 128 + 127;         // block causal limit
  const int qrow16 = w * 16 + lane16;        // A-fragment row within Qs

  for (int kb0 = 0; kb0 <= kend; kb0 += 32) {
    const int  cur  = (kb0 >> 5) & 1;
    const bool more = (kb0 + 32 <= kend);

    uint4 kreg, vreg;
    if (more) {                              // loads for next slab in flight
      kreg = *(const uint4*)(Kb + (size_t)(kb0 + 32 + krow) * DK + koff);
      vreg = *(const uint4*)(Vb + (size_t)vrow * S + kb0 + 32 + voff);
    }

    // ---- scores: fragments loaded from LDS just-in-time
    v8f c[2] = {};
#pragma unroll
    for (int ks = 0; ks < 2; ++ks) {
      ABReg aqf;
      *(uint4*)&aqf.u[0] = *(const uint4*)&Qs[qrow16 * 64 + ks * 32 + half * 8];
      *(uint4*)&aqf.u[4] = *(const uint4*)&Qs[qrow16 * 64 + ks * 32 + half * 8 + 16];
#pragma unroll
      for (int nt = 0; nt < 2; ++nt) {
        ABReg bk;
        int key = nt * 16 + lane16;
        *(uint4*)&bk.u[0] = *(const uint4*)&Ks[cur][key * 64 + ks * 32 + half * 8];
        *(uint4*)&bk.u[4] = *(const uint4*)&Ks[cur][key * 64 + ks * 32 + half * 8 + 16];
        c[nt] = wmma_bf16(aqf, bk, c[nt]);
      }
    }

    // ---- scale, causal mask, online softmax (row = VGPR slot r + 8*half)
#pragma unroll
    for (int r = 0; r < 8; ++r) {
      int qrow = q0 + r + 8 * half;
      float s0 = c[0][r] * SCALE;
      float s1 = c[1][r] * SCALE;
      if (kb0 + lane16 > qrow)      s0 = -1e30f;
      if (kb0 + 16 + lane16 > qrow) s1 = -1e30f;
      float tmax = fmaxf(s0, s1);
#pragma unroll
      for (int d = 1; d < 16; d <<= 1) tmax = fmaxf(tmax, __shfl_xor(tmax, d, 32));
      float mnew  = fmaxf(mrow[r], tmax);
      float alpha = __expf(mrow[r] - mnew);
      float p0 = __expf(s0 - mnew);
      float p1 = __expf(s1 - mnew);
      float rsum = p0 + p1;
#pragma unroll
      for (int d = 1; d < 16; d <<= 1) rsum += __shfl_xor(rsum, d, 32);
      lrow[r] = lrow[r] * alpha + rsum;
      mrow[r] = mnew;
#pragma unroll
      for (int t = 0; t < 4; ++t) o[t][r] *= alpha;
      pbuf[w][(r + 8 * half) * 32 + lane16]      = f2bf(p0);
      pbuf[w][(r + 8 * half) * 32 + 16 + lane16] = f2bf(p1);
    }
    asm volatile("s_wait_dscnt 0" ::: "memory");

    // ---- reload P as a 16x32 A-tile
    ABReg ap;
    *(uint4*)&ap.u[0] = *(const uint4*)&pbuf[w][lane16 * 32 + half * 8];
    *(uint4*)&ap.u[4] = *(const uint4*)&pbuf[w][lane16 * 32 + half * 8 + 16];

    // ---- O += P @ V   (V fragments loaded just-in-time)
#pragma unroll
    for (int t = 0; t < 4; ++t) {
      ABReg bv;
      int dk = t * 16 + lane16;
      *(uint4*)&bv.u[0] = *(const uint4*)&Vs[cur][dk * 32 + half * 8];
      *(uint4*)&bv.u[4] = *(const uint4*)&Vs[cur][dk * 32 + half * 8 + 16];
      o[t] = wmma_bf16(ap, bv, o[t]);
    }

    if (more) {                              // fill the other buffer
      *(uint4*)&Ks[cur ^ 1][krow * 64 + koff] = kreg;
      *(uint4*)&Vs[cur ^ 1][vrow * 32 + voff] = vreg;
    }
    __syncthreads();
  }

  // ---- epilogue: O / l, write bf16 [B,S,D]
#pragma unroll
  for (int t = 0; t < 4; ++t)
#pragma unroll
    for (int r = 0; r < 8; ++r) {
      int srow = q0 + r + 8 * half;
      float v = o[t][r] / lrow[r];
      O[((size_t)(b * S + srow)) * D + h * DK + t * 16 + lane16] = f2bf(v);
    }
}

// ---------------------------------------------------------------- launch
extern "C" void kernel_launch(void* const* d_in, const int* in_sizes, int n_in,
                              void* d_out, int out_size, void* d_ws, size_t ws_size,
                              hipStream_t stream) {
  const float* x  = (const float*)d_in[0];
  const int*   tp = (const int*)d_in[1];
  const float* wq = (const float*)d_in[2];
  const float* wk = (const float*)d_in[3];
  const float* wv = (const float*)d_in[4];
  const float* wo = (const float*)d_in[5];

  char* ws = (char*)d_ws;
  unsigned short* xb  = (unsigned short*)(ws);                       // 8 MB  x bf16
  unsigned short* wqb = (unsigned short*)(ws + (size_t)(8  << 20));
  unsigned short* wkb = (unsigned short*)(ws + (size_t)(10 << 20));
  unsigned short* wvb = (unsigned short*)(ws + (size_t)(12 << 20));
  unsigned short* wob = (unsigned short*)(ws + (size_t)(14 << 20));
  unsigned short* qb  = (unsigned short*)(ws + (size_t)(16 << 20));  // [B,H,S,DK]
  unsigned short* kbf = (unsigned short*)(ws + (size_t)(24 << 20));  // [B,H,S,DK]
  unsigned short* vtb = (unsigned short*)(ws + (size_t)(32 << 20));  // [B,H,DK,S]
  unsigned short* ob  = (unsigned short*)(ws + (size_t)(40 << 20));  // [B,S,D]

  convert_f32_bf16<<<1024, 256, 0, stream>>>(x,  xb,  M * D);
  convert_f32_bf16<<<512,  256, 0, stream>>>(wq, wqb, D * D);
  convert_f32_bf16<<<512,  256, 0, stream>>>(wk, wkb, D * D);
  convert_f32_bf16<<<512,  256, 0, stream>>>(wv, wvb, D * D);
  convert_f32_bf16<<<512,  256, 0, stream>>>(wo, wob, D * D);

  // (4096/128) x (1024/128) = 256 blocks of 8 waves
  gemm_bf16_wmma<<<256, 256, 0, stream>>>(xb, wqb, qb,  0);  // Q
  gemm_bf16_wmma<<<256, 256, 0, stream>>>(xb, wkb, kbf, 0);  // K
  gemm_bf16_wmma<<<256, 256, 0, stream>>>(xb, wvb, vtb, 1);  // V (transposed)

  rope_kernel<<<(B * H * S * (DK / 2)) / 256, 256, 0, stream>>>(qb, kbf, tp);

  // B*H*(S/128) = 512 blocks, each 8 waves / 128 query rows
  attn_kernel<<<512, 256, 0, stream>>>(qb, kbf, vtb, ob);

  gemm_bf16_wmma<<<256, 256, 0, stream>>>(ob, wob, d_out, 2); // output projection
}
